// Stag_721554506362
// MI455X (gfx1250) — compile-verified, matrix operand order checked by reading, not run on hardware
//
#include <hip/hip_runtime.h>
#include <hip/hip_bf16.h>

#define N_NODES 50000
#define N_EDGES 1600000
#define IN_FEAT 128
#define HEADS 8
#define OUT_FEAT 16

// LDS row stride (halves): 128 + 8 pad -> 272B/row = 68 dwords, so lane i's
// b128 lands on banks 4i..4i+3 (mod 64): conflict-free across 16 lanes.
#define LSTR 136

typedef __attribute__((ext_vector_type(16))) _Float16 v16h;
typedef __attribute__((ext_vector_type(8)))  _Float16 v8h;
typedef __attribute__((ext_vector_type(8)))  float    v8f;

// ---------------------------------------------------------------------------
// Phase A: h = feat(50000x128) @ W_fc(128x128), f16 WMMA with f32 accum.
// Block = 256 threads = 8 waves. All waves share rowTile = blockIdx.x;
// wave w computes colTile w. A tile (16x128) and a transposed f16 copy of
// the whole W_fc (128x128) are staged in LDS once per block; the WMMA inner
// loop is fed purely by ds_load_b128.
//
// A layout (16-bit A 16x32): lane&15 = row M; kbase = 8*(lane>>4);
//   halves a[0..7]  = K k0..k0+7   (contiguous)
//   halves a[8..15] = K k0+16..k0+23 (contiguous)  -> two ds_load_b128.
// B mirrors A with lane&15 = column N (B stored transposed in LDS).
// D layout: lane&15 = N, VGPR r: M = 8*(lane>>4) + r.
// ---------------------------------------------------------------------------
__global__ void gat_gemm_wmma(const float* __restrict__ feat,
                              const float* __restrict__ Wfc,
                              float* __restrict__ h) {
  __shared__ _Float16 Alds[16 * LSTR];    //  4.25 KB, [m][k]
  __shared__ _Float16 Blds[128 * LSTR];   // 34 KB,   [n][k] (transposed W_fc)

  const int rowTile = (int)blockIdx.x;    // 0..3124
  const int tid = (int)threadIdx.x;

  // Stage A tile: 16x128 f32, coalesced reads, f16 convert.
  {
    const float* __restrict__ abase = feat + (long)rowTile * 16 * IN_FEAT;
    #pragma unroll
    for (int t = tid; t < 16 * 128; t += 256) {
      const int m = t >> 7, k = t & 127;
      Alds[m * LSTR + k] = (_Float16)abase[t];
    }
  }
  // Stage W_fc transposed: read row-major coalesced, write [n][k].
  #pragma unroll
  for (int t = tid; t < 128 * 128; t += 256) {
    const int k = t >> 7, n = t & 127;
    Blds[n * LSTR + k] = (_Float16)Wfc[t];
  }
  __syncthreads();

  const int lane = tid & 31;
  const int colTile = tid >> 5;           // 0..7
  const int half = lane >> 4;
  const int lidx = lane & 15;
  const int kbase = half * 8;

  const _Float16* Ab = &Alds[lidx * LSTR];
  const _Float16* Bb = &Blds[(colTile * 16 + lidx) * LSTR];

  v8f c = {};
  #pragma unroll
  for (int s = 0; s < 4; ++s) {
    const int k0 = s * 32 + kbase;
    const v8h a0 = *(const v8h*)(Ab + k0);
    const v8h a1 = *(const v8h*)(Ab + k0 + 16);
    const v8h b0 = *(const v8h*)(Bb + k0);
    const v8h b1 = *(const v8h*)(Bb + k0 + 16);
    const v16h a = __builtin_shufflevector(a0, a1, 0, 1, 2, 3, 4, 5, 6, 7,
                                           8, 9, 10, 11, 12, 13, 14, 15);
    const v16h b = __builtin_shufflevector(b0, b1, 0, 1, 2, 3, 4, 5, 6, 7,
                                           8, 9, 10, 11, 12, 13, 14, 15);
    c = __builtin_amdgcn_wmma_f32_16x16x32_f16(false, a, false, b,
                                               (short)0, c, false, false);
  }

  const int outcol = colTile * 16 + lidx;
  #pragma unroll
  for (int r = 0; r < 8; ++r) {
    const int m = rowTile * 16 + half * 8 + r;
    h[(long)m * 128 + outcol] = c[r];
  }
}

// ---------------------------------------------------------------------------
// Init: out = bias broadcast; maxkey = 0 (== -inf key); denom = 0
// ---------------------------------------------------------------------------
__global__ void gat_init(float* __restrict__ out, const float* __restrict__ bias,
                         unsigned* __restrict__ maxkey, float* __restrict__ denom) {
  const int idx = (int)(blockIdx.x * blockDim.x + threadIdx.x);
  if (idx < N_NODES * HEADS * OUT_FEAT) out[idx] = bias[idx & 127];
  if (idx < N_NODES * HEADS) { maxkey[idx] = 0u; denom[idx] = 0.0f; }
}

// ---------------------------------------------------------------------------
// Phase B: post[n,h,0:4] = h[n,h,:] @ W_post(16x4) + b_post
// ---------------------------------------------------------------------------
__global__ void gat_post(const float* __restrict__ h,
                         const float* __restrict__ Wpost,
                         const float* __restrict__ bpost,
                         float* __restrict__ post) {
  const int idx = (int)(blockIdx.x * blockDim.x + threadIdx.x);
  if (idx >= N_NODES * HEADS) return;
  const float* __restrict__ hv = h + (long)idx * OUT_FEAT;
  float acc0 = bpost[0], acc1 = bpost[1], acc2 = bpost[2], acc3 = bpost[3];
  #pragma unroll
  for (int f = 0; f < OUT_FEAT; ++f) {
    const float x = hv[f];
    acc0 = fmaf(x, Wpost[f * 4 + 0], acc0);
    acc1 = fmaf(x, Wpost[f * 4 + 1], acc1);
    acc2 = fmaf(x, Wpost[f * 4 + 2], acc2);
    acc3 = fmaf(x, Wpost[f * 4 + 3], acc3);
  }
  float* p = post + (long)idx * 4;
  p[0] = acc0; p[1] = acc1; p[2] = acc2; p[3] = acc3;
}

// Monotonic uint key for float atomic-max
__device__ __forceinline__ unsigned fkey(float f) {
  unsigned b = __float_as_uint(f);
  return b ^ ((b & 0x80000000u) ? 0xFFFFFFFFu : 0x80000000u);
}
__device__ __forceinline__ float fkeyinv(unsigned k) {
  unsigned b = k ^ ((k & 0x80000000u) ? 0x80000000u : 0xFFFFFFFFu);
  return __uint_as_float(b);
}

__device__ __forceinline__ float edge_e(const float* __restrict__ post,
                                        const float* __restrict__ eps,
                                        int s, int d, int hd, int eid) {
  const float* ps = post + ((long)s * HEADS + hd) * 4;
  const float* pd = post + ((long)d * HEADS + hd) * 4;
  const float loc = ps[0] + pd[1];
  const float ls  = ps[2] + pd[3];
  return fmaf(__expf(ls), eps[(long)eid * HEADS + hd], loc);
}

// ---------------------------------------------------------------------------
// Phase C1: per-(edge,head) segment max over dst
// ---------------------------------------------------------------------------
__global__ void gat_edge_max(const int* __restrict__ src, const int* __restrict__ dst,
                             const float* __restrict__ eps, const float* __restrict__ post,
                             unsigned* __restrict__ maxkey) {
  const int idx = (int)(blockIdx.x * blockDim.x + threadIdx.x);
  if (idx >= N_EDGES * HEADS) return;
  const int eid = idx >> 3, hd = idx & 7;
  const int s = src[eid], d = dst[eid];
  const float e = edge_e(post, eps, s, d, hd, eid);
  atomicMax(&maxkey[(long)d * HEADS + hd], fkey(e));
}

// ---------------------------------------------------------------------------
// Phase C2: denom = segment_sum(exp(e - m))
// ---------------------------------------------------------------------------
__global__ void gat_edge_denom(const int* __restrict__ src, const int* __restrict__ dst,
                               const float* __restrict__ eps, const float* __restrict__ post,
                               const unsigned* __restrict__ maxkey,
                               float* __restrict__ denom) {
  const int idx = (int)(blockIdx.x * blockDim.x + threadIdx.x);
  if (idx >= N_EDGES * HEADS) return;
  const int eid = idx >> 3, hd = idx & 7;
  const int s = src[eid], d = dst[eid];
  const float e = edge_e(post, eps, s, d, hd, eid);
  const int didx = d * HEADS + hd;
  const float m = fkeyinv(maxkey[didx]);
  atomicAdd(&denom[didx], __expf(e - m));
}

// ---------------------------------------------------------------------------
// Phase C3: out[dst] += h[src] * a   (a = exp(e-m)/denom)
// ---------------------------------------------------------------------------
__global__ void gat_edge_agg(const int* __restrict__ src, const int* __restrict__ dst,
                             const float* __restrict__ eps, const float* __restrict__ post,
                             const float* __restrict__ hfeat,
                             const unsigned* __restrict__ maxkey,
                             const float* __restrict__ denom,
                             float* __restrict__ out) {
  const int idx = (int)(blockIdx.x * blockDim.x + threadIdx.x);
  if (idx >= N_EDGES * HEADS) return;
  const int eid = idx >> 3, hd = idx & 7;
  const int s = src[eid], d = dst[eid];
  const float e = edge_e(post, eps, s, d, hd, eid);
  const int didx = d * HEADS + hd;
  const float m = fkeyinv(maxkey[didx]);
  const float a = __expf(e - m) / denom[didx];
  const float4* __restrict__ hv = (const float4*)(hfeat + ((long)s * HEADS + hd) * OUT_FEAT);
  float* op = out + (long)didx * OUT_FEAT;
  #pragma unroll
  for (int q = 0; q < 4; ++q) {
    const float4 v = hv[q];
    atomicAdd(op + 4 * q + 0, v.x * a);
    atomicAdd(op + 4 * q + 1, v.y * a);
    atomicAdd(op + 4 * q + 2, v.z * a);
    atomicAdd(op + 4 * q + 3, v.w * a);
  }
}

extern "C" void kernel_launch(void* const* d_in, const int* in_sizes, int n_in,
                              void* d_out, int out_size, void* d_ws, size_t ws_size,
                              hipStream_t stream) {
  const float* feat  = (const float*)d_in[0];
  const int*   src   = (const int*)  d_in[1];
  const int*   dst   = (const int*)  d_in[2];
  const float* eps   = (const float*)d_in[3];
  const float* Wfc   = (const float*)d_in[4];
  const float* Wpost = (const float*)d_in[5];
  const float* bpost = (const float*)d_in[6];
  const float* bias  = (const float*)d_in[7];
  float* out = (float*)d_out;

  // Workspace carving (35.2 MB total)
  float*    h      = (float*)d_ws;                                    // 6,400,000 f32
  float*    post   = h + (size_t)N_NODES * HEADS * OUT_FEAT;          // 1,600,000 f32
  unsigned* maxkey = (unsigned*)(post + (size_t)N_NODES * HEADS * 4); //   400,000 u32
  float*    denom  = (float*)(maxkey + (size_t)N_NODES * HEADS);      //   400,000 f32

  const int BLK = 256;

  // A: WMMA GEMM — one block (8 waves) per 16-row tile, LDS-staged operands.
  gat_gemm_wmma<<<N_NODES / 16, BLK, 0, stream>>>(feat, Wfc, h);

  // Init out/maxkey/denom
  gat_init<<<(N_NODES * HEADS * OUT_FEAT + BLK - 1) / BLK, BLK, 0, stream>>>(out, bias, maxkey, denom);

  // B: posterior params
  gat_post<<<(N_NODES * HEADS + BLK - 1) / BLK, BLK, 0, stream>>>(h, Wpost, bpost, post);

  // C: edge softmax + aggregation
  const int egrid = (N_EDGES * HEADS + BLK - 1) / BLK;
  gat_edge_max  <<<egrid, BLK, 0, stream>>>(src, dst, eps, post, maxkey);
  gat_edge_denom<<<egrid, BLK, 0, stream>>>(src, dst, eps, post, maxkey, denom);
  gat_edge_agg  <<<egrid, BLK, 0, stream>>>(src, dst, eps, post, h, maxkey, denom, out);
}
